// SwavVQDisentangle_38474317038313
// MI455X (gfx1250) — matrix operand order repaired and checked by reading, not compiled, not run on hardware
//
#include <hip/hip_runtime.h>
#include <hip/hip_bf16.h>

// ---------------------------------------------------------------------------
// CDNA5 / gfx1250 implementation of the SwAV-VQ disentangle loss.
// GEMMs (logits = zn @ W^T, Gram = z^T z) run on v_wmma_f32_16x16x32_bf16.
// Sinkhorn is kept factorized: Q = exp(l/eps) * u[k] * v[b], only the scaling
// vectors are updated per iteration (row pass -> u, col pass -> v).
// ---------------------------------------------------------------------------

typedef __attribute__((ext_vector_type(16))) __bf16 v16bf;
typedef __attribute__((ext_vector_type(8)))  __bf16 v8bf;
typedef __attribute__((ext_vector_type(8)))  float  v8f;

static constexpr int   Bn      = 16384;
static constexpr int   Dn      = 512;
static constexpr int   Kn      = 2048;
static constexpr float EPS_    = 0.05f;   // sinkhorn epsilon
static constexpr float TEMP_   = 0.1f;
static constexpr float LAMBD_  = 15.0f;
static constexpr float MU_     = 1.0f;
static constexpr float NU_     = 1.0f;
static constexpr float GAMMA_  = 0.2f;

__device__ __forceinline__ float waveRedSum(float v) {
#pragma unroll
  for (int o = 16; o > 0; o >>= 1) v += __shfl_xor(v, o, 32);
  return v;
}

// ---------------- small utility kernels ------------------------------------

__global__ void k_zero(float* __restrict__ p, int n) {
  int i = blockIdx.x * 256 + threadIdx.x;
  if (i < n) p[i] = 0.f;
}

// per-column sum / sumsq partials (coalesced: thread <-> column)
__global__ void k_colstats(const float* __restrict__ z,
                           float* __restrict__ colsum, float* __restrict__ colsq) {
  int d  = blockIdx.x * 256 + threadIdx.x;   // gridDim.x = Dn/256
  int r0 = blockIdx.y * 256;                 // gridDim.y = Bn/256
  float s = 0.f, q = 0.f;
  for (int r = 0; r < 256; ++r) {
    float x = z[(size_t)(r0 + r) * Dn + d];
    s += x; q += x * x;
  }
  atomicAdd(&colsum[d], s);
  atomicAdd(&colsq[d], q);
}

// finalize: mean[d], unbiased var -> hinge variance loss accumulation
__global__ void k_colfinal(const float* __restrict__ colsum, const float* __restrict__ colsq,
                           float* __restrict__ mean, float* __restrict__ vsum) {
  int d = blockIdx.x * 256 + threadIdx.x;
  if (d >= Dn) return;
  float m   = colsum[d] * (1.f / (float)Bn);
  float var = (colsq[d] - (float)Bn * m * m) * (1.f / (float)(Bn - 1));
  mean[d] = m;
  float t = GAMMA_ - sqrtf(var + 1e-8f);
  float r = t > 0.f ? t : 0.f;
  r = waveRedSum(r);
  if ((threadIdx.x & 31) == 0) atomicAdd(vsum, r);
}

// z [B,D] f32 -> zT [D,B] bf16 (for the Gram GEMM: inner dim must be contiguous)
__global__ void k_transpose_bf16(const float* __restrict__ z, __bf16* __restrict__ zT) {
  size_t i = (size_t)blockIdx.x * 256 + threadIdx.x;   // Bn*Dn total
  int b = (int)(i / Dn), d = (int)(i % Dn);
  zT[(size_t)d * Bn + b] = (__bf16)z[i];
}

// L2-normalize each row of z, emit bf16 (wave per row)
__global__ void k_rownorm_bf16(const float* __restrict__ z, __bf16* __restrict__ zn) {
  int wave = threadIdx.x >> 5, lane = threadIdx.x & 31;
  int b = blockIdx.x * 8 + wave;
  const float* row = z + (size_t)b * Dn;
  float s = 0.f;
  for (int k = lane; k < Dn; k += 32) { float x = row[k]; s += x * x; }
  s = waveRedSum(s);
  float nrm = sqrtf(s);
  float inv = 1.f / (nrm > 1e-12f ? nrm : 1e-12f);
  __bf16* out = zn + (size_t)b * Dn;
  for (int k = lane; k < Dn; k += 32) out[k] = (__bf16)(row[k] * inv);
}

__global__ void k_cvt_bf16(const float* __restrict__ x, __bf16* __restrict__ y, int n) {
  int i = blockIdx.x * 256 + threadIdx.x;
  if (i < n) y[i] = (__bf16)x[i];
}

// ---------------- WMMA GEMM:  C[M,N] = A[M,Kd] * Bm[N,Kd]^T -----------------
// block = 256 threads = 8 waves arranged 2(M) x 4(N); wave tile = 32x32
// (2x2 v_wmma_f32_16x16x32_bf16). mode 0: store C. mode 1: covariance
// epilogue (C_ij = (G - n m_i m_j)/(n-1), sum of squared off-diagonals).
__global__ void __launch_bounds__(256)
k_gemm_bf16(const __bf16* __restrict__ A, const __bf16* __restrict__ Bm,
            float* __restrict__ C, int M, int N, int Kd,
            int mode, const float* __restrict__ mean, float* __restrict__ cov_accum) {
  int tid  = threadIdx.x;
  int wave = tid >> 5, lane = tid & 31;
  int half = lane >> 4, l16 = lane & 15;
  int wm0 = blockIdx.x * 64  + (wave >> 2) * 32;
  int wn0 = blockIdx.y * 128 + (wave & 3) * 32;
  (void)M;

  v8f c00 = {}, c01 = {}, c10 = {}, c11 = {};

  for (int k0 = 0; k0 < Kd; k0 += 32) {
    // A fragment, 16x32 bf16 per ISA layout:
    //   lane<16: row=l16, K = {k0..k0+7} in v0-3, {k0+16..k0+23} in v4-7
    //   lane>=16: same row set, K shifted by +8
    const __bf16* a0p = A + (size_t)(wm0 + l16) * Kd + k0 + half * 8;
    const __bf16* a1p = A + (size_t)(wm0 + 16 + l16) * Kd + k0 + half * 8;
    v8bf a0lo = *(const v8bf*)a0p;        v8bf a0hi = *(const v8bf*)(a0p + 16);
    v8bf a1lo = *(const v8bf*)a1p;        v8bf a1hi = *(const v8bf*)(a1p + 16);
    v16bf a0 = __builtin_shufflevector(a0lo, a0hi, 0,1,2,3,4,5,6,7,8,9,10,11,12,13,14,15);
    v16bf a1 = __builtin_shufflevector(a1lo, a1hi, 0,1,2,3,4,5,6,7,8,9,10,11,12,13,14,15);
    // B fragment, 32x16: lane = output column n; same K order as A:
    // lane<16 holds K {k0..k0+7, k0+16..k0+23}, lane>=16 shifted by +8
    const __bf16* b0p = Bm + (size_t)(wn0 + l16) * Kd + k0 + half * 8;
    const __bf16* b1p = Bm + (size_t)(wn0 + 16 + l16) * Kd + k0 + half * 8;
    v8bf b0lo = *(const v8bf*)b0p;        v8bf b0hi = *(const v8bf*)(b0p + 16);
    v8bf b1lo = *(const v8bf*)b1p;        v8bf b1hi = *(const v8bf*)(b1p + 16);
    v16bf b0 = __builtin_shufflevector(b0lo, b0hi, 0,1,2,3,4,5,6,7,8,9,10,11,12,13,14,15);
    v16bf b1 = __builtin_shufflevector(b1lo, b1hi, 0,1,2,3,4,5,6,7,8,9,10,11,12,13,14,15);

    c00 = __builtin_amdgcn_wmma_f32_16x16x32_bf16(false, a0, false, b0, (short)0, c00, false, false);
    c01 = __builtin_amdgcn_wmma_f32_16x16x32_bf16(false, a0, false, b1, (short)0, c01, false, false);
    c10 = __builtin_amdgcn_wmma_f32_16x16x32_bf16(false, a1, false, b0, (short)0, c10, false, false);
    c11 = __builtin_amdgcn_wmma_f32_16x16x32_bf16(false, a1, false, b1, (short)0, c11, false, false);
  }

  if (mode == 0) {
#pragma unroll
    for (int r = 0; r < 8; ++r) {
      int gm = half * 8 + r;   // C/D layout: VGPR r -> M = r (+8 for lanes 16-31)
      C[(size_t)(wm0 + gm)      * N + wn0 + l16]      = c00[r];
      C[(size_t)(wm0 + gm)      * N + wn0 + 16 + l16] = c01[r];
      C[(size_t)(wm0 + 16 + gm) * N + wn0 + l16]      = c10[r];
      C[(size_t)(wm0 + 16 + gm) * N + wn0 + 16 + l16] = c11[r];
    }
  } else {
    float inv = 1.f / (float)(Bn - 1);
    float s = 0.f;
#pragma unroll
    for (int r = 0; r < 8; ++r) {
      int gmA = wm0 + half * 8 + r, gmB = gmA + 16;
      int gnA = wn0 + l16,          gnB = gnA + 16;
      float v;
      v = (c00[r] - (float)Bn * mean[gmA] * mean[gnA]) * inv; if (gmA != gnA) s += v * v;
      v = (c01[r] - (float)Bn * mean[gmA] * mean[gnB]) * inv; if (gmA != gnB) s += v * v;
      v = (c10[r] - (float)Bn * mean[gmB] * mean[gnA]) * inv; if (gmB != gnA) s += v * v;
      v = (c11[r] - (float)Bn * mean[gmB] * mean[gnB]) * inv; if (gmB != gnB) s += v * v;
    }
    s = waveRedSum(s);
    if (lane == 0) atomicAdd(cov_accum, s);
  }
}

// ---------------- Sinkhorn (factorized) -------------------------------------
// column-sum over b for each k: T[k] += sum_b exp(l/eps) * v[b]
__global__ void k_rowstep(const float* __restrict__ logits, const float* __restrict__ v,
                          float* __restrict__ T, int use_v) {
  int k  = blockIdx.x * 256 + threadIdx.x;  // gridDim.x = Kn/256
  int r0 = blockIdx.y * 128;                // gridDim.y = Bn/128
  float s = 0.f;
  for (int r = 0; r < 128; ++r) {
    int b = r0 + r;
    float e = __expf(logits[(size_t)b * Kn + k] * (1.f / EPS_));
    s += use_v ? e * v[b] : e;
  }
  atomicAdd(&T[k], s);
}

__global__ void k_ufinal(const float* __restrict__ T, float* __restrict__ u,
                         float* __restrict__ logu) {
  int k = blockIdx.x * 256 + threadIdx.x;
  if (k >= Kn) return;
  float uu = 1.f / ((float)Kn * T[k]);
  u[k] = uu; logu[k] = __logf(uu);
}

// row-sum over k for each b (wave per row): v[b] = 1/(B * sum_k exp(l/eps)*u[k])
__global__ void k_colstep(const float* __restrict__ logits, const float* __restrict__ u,
                          float* __restrict__ v) {
  int wave = threadIdx.x >> 5, lane = threadIdx.x & 31;
  int b = blockIdx.x * 8 + wave;
  const float* row = logits + (size_t)b * Kn;
  float s = 0.f;
  for (int k = lane; k < Kn; k += 32) s += __expf(row[k] * (1.f / EPS_)) * u[k];
  s = waveRedSum(s);
  if (lane == 0) v[b] = 1.f / ((float)Bn * s);
}

// ---------------- per-row stats: max, LSE(l/T), argmax(l), argmax(tgt) ------
__global__ void k_rowstats(const float* __restrict__ logits, const float* __restrict__ logu,
                           float* __restrict__ rmax, float* __restrict__ rlse,
                           int* __restrict__ amax_l, int* __restrict__ amax_t,
                           float* __restrict__ hard_hist) {
  int wave = threadIdx.x >> 5, lane = threadIdx.x & 31;
  int b = blockIdx.x * 8 + wave;
  const float* row = logits + (size_t)b * Kn;
  float mv = -3.4e38f; int mi = 0;    // argmax of raw logits (== argmax of l/T)
  float tv = -3.4e38f; int ti = 0;    // argmax of l/eps + log u  (== argmax tgt)
  for (int k = lane; k < Kn; k += 32) {
    float x = row[k];
    if (x > mv) { mv = x; mi = k; }
    float t = x * (1.f / EPS_) + logu[k];
    if (t > tv) { tv = t; ti = k; }
  }
#pragma unroll
  for (int o = 16; o > 0; o >>= 1) {
    float omv = __shfl_xor(mv, o, 32); int omi = __shfl_xor(mi, o, 32);
    if (omv > mv || (omv == mv && omi < mi)) { mv = omv; mi = omi; }
    float otv = __shfl_xor(tv, o, 32); int oti = __shfl_xor(ti, o, 32);
    if (otv > tv || (otv == tv && oti < ti)) { tv = otv; ti = oti; }
  }
  float ms = mv * (1.f / TEMP_);
  float s = 0.f;
  for (int k = lane; k < Kn; k += 32) s += __expf(row[k] * (1.f / TEMP_) - ms);
  s = waveRedSum(s);
  if (lane == 0) {
    rmax[b] = ms; rlse[b] = __logf(s);
    amax_l[b] = mi; amax_t[b] = ti;
    atomicAdd(&hard_hist[mi], 1.f);
  }
}

__global__ void k_acc(const int* __restrict__ al1, const int* __restrict__ at2,
                      const int* __restrict__ al2, const int* __restrict__ at1,
                      float* __restrict__ acc_sum) {
  int b = blockIdx.x * 256 + threadIdx.x;
  float a = (al1[b] == at2[b] ? 1.f : 0.f) + (al2[b] == at1[b] ? 1.f : 0.f);
  a = waveRedSum(a);
  if ((threadIdx.x & 31) == 0) atomicAdd(acc_sum, a);
}

// ---------------- fused CE + avg-probs pass ---------------------------------
__global__ void k_ce(const float* __restrict__ l1, const float* __restrict__ l2,
                     const float* __restrict__ u1, const float* __restrict__ v1,
                     const float* __restrict__ u2, const float* __restrict__ v2,
                     const float* __restrict__ m1, const float* __restrict__ s1,
                     const float* __restrict__ m2, const float* __restrict__ s2,
                     float* __restrict__ ce_sum, float* __restrict__ avg_probs) {
  __shared__ float sAvg[Kn];
  for (int k = threadIdx.x; k < Kn; k += 256) sAvg[k] = 0.f;
  __syncthreads();
  int wave = threadIdx.x >> 5, lane = threadIdx.x & 31;
  int b = blockIdx.x * 8 + wave;
  const float* r1 = l1 + (size_t)b * Kn;
  const float* r2 = l2 + (size_t)b * Kn;
  float base1 = m1[b] + s1[b], base2 = m2[b] + s2[b];
  float vb1 = v1[b] * (float)Bn, vb2 = v2[b] * (float)Bn;
  float ce = 0.f;
  for (int k = lane; k < Kn; k += 32) {
    float a1 = r1[k], a2 = r2[k];
    float lp1 = a1 * (1.f / TEMP_) - base1;
    float lp2 = a2 * (1.f / TEMP_) - base2;
    float t1 = __expf(a1 * (1.f / EPS_)) * u1[k] * vb1;   // tgt_probs_1[b,k]
    float t2 = __expf(a2 * (1.f / EPS_)) * u2[k] * vb2;   // tgt_probs_2[b,k]
    ce += t1 * lp2 + t2 * lp1;
    atomicAdd(&sAvg[k], __expf(lp1) + __expf(lp2));       // softmax rows, both views
  }
  ce = waveRedSum(ce);
  if (lane == 0) atomicAdd(ce_sum, ce);
  __syncthreads();
  for (int k = threadIdx.x; k < Kn; k += 256) atomicAdd(&avg_probs[k], sAvg[k]);
}

// ---------------- final scalar assembly -------------------------------------
__global__ void k_final(const float* __restrict__ vsum, const float* __restrict__ covsum,
                        const float* __restrict__ ce_sum, const float* __restrict__ acc_sum,
                        const float* __restrict__ hard_hist, const float* __restrict__ avg_probs,
                        float* __restrict__ out) {
  __shared__ float red[256];
  float hc = 0.f, hp = 0.f;
  for (int k = threadIdx.x; k < Kn; k += 256) {
    float h = hard_hist[k] * (1.f / (2.f * (float)Bn));
    hc += h * __logf(h + 1e-7f);
    float p = avg_probs[k] * (1.f / (2.f * (float)Bn));
    hp += p * __logf(p + 1e-7f);
  }
  red[threadIdx.x] = hc; __syncthreads();
  for (int s = 128; s > 0; s >>= 1) { if (threadIdx.x < s) red[threadIdx.x] += red[threadIdx.x + s]; __syncthreads(); }
  hc = red[0]; __syncthreads();
  red[threadIdx.x] = hp; __syncthreads();
  for (int s = 128; s > 0; s >>= 1) { if (threadIdx.x < s) red[threadIdx.x] += red[threadIdx.x + s]; __syncthreads(); }
  hp = red[0];
  if (threadIdx.x == 0) {
    float loss_var = vsum[0] * (1.f / (float)Dn);
    float loss_cov = covsum[0] * (1.f / (float)Dn);
    float loss_ce  = -0.5f / (float)Bn * ce_sum[0];
    out[0] = LAMBD_ * loss_ce + MU_ * loss_var + NU_ * loss_cov;
    out[1] = loss_ce;
    out[2] = loss_var;
    out[3] = loss_cov;
    out[4] = __expf(-hc);
    out[5] = __expf(-hp);
    out[6] = acc_sum[0] * (1.f / (2.f * (float)Bn));
  }
}

// ---------------------------------------------------------------------------

extern "C" void kernel_launch(void* const* d_in, const int* in_sizes, int n_in,
                              void* d_out, int out_size, void* d_ws, size_t ws_size,
                              hipStream_t stream) {
  (void)in_sizes; (void)n_in; (void)out_size; (void)ws_size;
  const float* z1 = (const float*)d_in[0];
  const float* z2 = (const float*)d_in[1];
  const float* W  = (const float*)d_in[2];
  float* out = (float*)d_out;

  char* ws = (char*)d_ws;
  size_t off = 0;
  auto alloc = [&](size_t bytes) -> void* {
    void* p = ws + off; off += (bytes + 255) & ~(size_t)255; return p;
  };

  float*  logits1 = (float*) alloc((size_t)Bn * Kn * 4);
  float*  logits2 = (float*) alloc((size_t)Bn * Kn * 4);
  __bf16* zT1     = (__bf16*)alloc((size_t)Dn * Bn * 2);
  __bf16* zT2     = (__bf16*)alloc((size_t)Dn * Bn * 2);
  __bf16* zn1     = (__bf16*)alloc((size_t)Bn * Dn * 2);
  __bf16* zn2     = (__bf16*)alloc((size_t)Bn * Dn * 2);
  __bf16* Wb      = (__bf16*)alloc((size_t)Kn * Dn * 2);
  float*  mean1   = (float*) alloc(Dn * 4);
  float*  mean2   = (float*) alloc(Dn * 4);
  // accumulators zeroed in one shot: [colsum1|colsq1|colsum2|colsq2|hist|avg|scalars]
  int acc_n = 4 * Dn + 2 * Kn + 8;
  float* accb    = (float*)alloc((size_t)acc_n * 4);
  float* colsum1 = accb;            float* colsq1 = accb + Dn;
  float* colsum2 = accb + 2 * Dn;   float* colsq2 = accb + 3 * Dn;
  float* hard_hist = accb + 4 * Dn;
  float* avg_probs = accb + 4 * Dn + Kn;
  float* vsum   = accb + 4 * Dn + 2 * Kn;
  float* covsum = vsum + 1;
  float* ce_sum = vsum + 2;
  float* acc_s  = vsum + 3;
  float* T     = (float*)alloc(Kn * 4);
  float* u1    = (float*)alloc(Kn * 4);
  float* logu1 = (float*)alloc(Kn * 4);
  float* u2    = (float*)alloc(Kn * 4);
  float* logu2 = (float*)alloc(Kn * 4);
  float* v1    = (float*)alloc(Bn * 4);
  float* v2    = (float*)alloc(Bn * 4);
  float* rmax1 = (float*)alloc(Bn * 4);
  float* rlse1 = (float*)alloc(Bn * 4);
  float* rmax2 = (float*)alloc(Bn * 4);
  float* rlse2 = (float*)alloc(Bn * 4);
  int*   al1   = (int*)  alloc(Bn * 4);
  int*   at1   = (int*)  alloc(Bn * 4);
  int*   al2   = (int*)  alloc(Bn * 4);
  int*   at2   = (int*)  alloc(Bn * 4);

  // 1) zero all accumulators
  k_zero<<<(acc_n + 255) / 256, 256, 0, stream>>>(accb, acc_n);

  // 2) column stats (mean/var) for both views
  dim3 gs(Dn / 256, Bn / 256);
  k_colstats<<<gs, 256, 0, stream>>>(z1, colsum1, colsq1);
  k_colstats<<<gs, 256, 0, stream>>>(z2, colsum2, colsq2);
  k_colfinal<<<Dn / 256, 256, 0, stream>>>(colsum1, colsq1, mean1, vsum);
  k_colfinal<<<Dn / 256, 256, 0, stream>>>(colsum2, colsq2, mean2, vsum);

  // 3) conversions: transposed bf16 (Gram), row-normalized bf16 (logits), W bf16
  int nz = Bn * Dn;
  k_transpose_bf16<<<nz / 256, 256, 0, stream>>>(z1, zT1);
  k_transpose_bf16<<<nz / 256, 256, 0, stream>>>(z2, zT2);
  k_rownorm_bf16<<<Bn / 8, 256, 0, stream>>>(z1, zn1);
  k_rownorm_bf16<<<Bn / 8, 256, 0, stream>>>(z2, zn2);
  k_cvt_bf16<<<(Kn * Dn) / 256, 256, 0, stream>>>(W, Wb, Kn * Dn);

  // 4) covariance Grams with fused off-diagonal reduction (WMMA)
  dim3 gg(Dn / 64, Dn / 128);
  k_gemm_bf16<<<gg, 256, 0, stream>>>(zT1, zT1, nullptr, Dn, Dn, Bn, 1, mean1, covsum);
  k_gemm_bf16<<<gg, 256, 0, stream>>>(zT2, zT2, nullptr, Dn, Dn, Bn, 1, mean2, covsum);

  // 5) logits GEMMs (WMMA)
  dim3 gl(Bn / 64, Kn / 128);
  k_gemm_bf16<<<gl, 256, 0, stream>>>(zn1, Wb, logits1, Bn, Kn, Dn, 0, nullptr, nullptr);
  k_gemm_bf16<<<gl, 256, 0, stream>>>(zn2, Wb, logits2, Bn, Kn, Dn, 0, nullptr, nullptr);

  // 6) sinkhorn (3 iterations of row-normalize then col-normalize, factorized)
  auto sinkhorn = [&](float* L, float* u, float* logu, float* v) {
    for (int it = 0; it < 3; ++it) {
      k_zero<<<(Kn + 255) / 256, 256, 0, stream>>>(T, Kn);
      k_rowstep<<<dim3(Kn / 256, Bn / 128), 256, 0, stream>>>(L, v, T, it > 0 ? 1 : 0);
      k_ufinal<<<(Kn + 255) / 256, 256, 0, stream>>>(T, u, logu);
      k_colstep<<<Bn / 8, 256, 0, stream>>>(L, u, v);
    }
  };
  sinkhorn(logits1, u1, logu1, v1);
  sinkhorn(logits2, u2, logu2, v2);

  // 7) per-row softmax stats, argmaxes, hard-code histogram
  k_rowstats<<<Bn / 8, 256, 0, stream>>>(logits1, logu1, rmax1, rlse1, al1, at1, hard_hist);
  k_rowstats<<<Bn / 8, 256, 0, stream>>>(logits2, logu2, rmax2, rlse2, al2, at2, hard_hist);
  k_acc<<<Bn / 256, 256, 0, stream>>>(al1, at2, al2, at1, acc_s);

  // 8) fused CE + avg-probs pass
  k_ce<<<Bn / 8, 256, 0, stream>>>(logits1, logits2, u1, v1, u2, v2,
                                   rmax1, rlse1, rmax2, rlse2, ce_sum, avg_probs);

  // 9) assemble the 7 scalar outputs
  k_final<<<1, 256, 0, stream>>>(vsum, covsum, ce_sum, acc_s, hard_hist, avg_probs, out);
}
